// VanillaRNN_55216099557906
// MI455X (gfx1250) — matrix operand order repaired and checked by reading, not compiled, not run on hardware
//
#include <hip/hip_runtime.h>
#include <hip/hip_bf16.h>
#include <stdint.h>

typedef __attribute__((ext_vector_type(16))) __bf16       bf16x16;
typedef __attribute__((ext_vector_type(8)))  float        f32x8;
typedef __attribute__((ext_vector_type(8)))  unsigned int u32x8;

#define B_ 256
#define T_ 1024
#define H_ 512
#define I_ 256
#define O_ 256
#define RS 520   // LDS row stride (bf16 elems): 16B aligned, bank-conflict free
#define KRES 4   // k-chunks of Whh kept resident in VGPRs across all timesteps

static __device__ __forceinline__ f32x8 wmma_bf16(bf16x16 a, bf16x16 b, f32x8 c) {
    return __builtin_amdgcn_wmma_f32_16x16x32_bf16(false, a, false, b, (short)0, c, false, false);
}

static __device__ __forceinline__ float fast_tanh(float x) {
#if __has_builtin(__builtin_amdgcn_tanhf)
    return __builtin_amdgcn_tanhf(x);          // v_tanh_f32 (gfx1250 trans op)
#else
    float e = __expf(2.0f * x);                // tanh = 1 - 2/(e^{2x}+1)
    return 1.0f - 2.0f * __builtin_amdgcn_rcpf(e + 1.0f);
#endif
}

// Load a 16x32 bf16 A-fragment from row-major LDS h (row stride RS elems).
// lanes 0-15: M=lane, K = kc*32 + {0..7, 16..23}; lanes 16-31: K = {8..15, 24..31}
static __device__ __forceinline__ bf16x16 load_a_lds(const __bf16* p) {
    uint4 a0 = *(const uint4*)p;        // K offs 0..7  (VGPR 0-3)
    uint4 a1 = *(const uint4*)(p + 16); // K offs 16..23 (VGPR 4-7)
    u32x8 t;
    t[0]=a0.x; t[1]=a0.y; t[2]=a0.z; t[3]=a0.w;
    t[4]=a1.x; t[5]=a1.y; t[6]=a1.z; t[7]=a1.w;
    return __builtin_bit_cast(bf16x16, t);
}

// ---------- repack W[N,K] f32 row-major -> WMMA B-fragment-ordered bf16 ----------
// frag(j,c): dword index = ((j*KC + c)*32 + lane)*8 + d
// lane n = j*16+(lane&15); K = c*32 + (lane>=16 ? 8:0) + 2d + (d>=4 ? 8:0) (+1 in high half)
__global__ void repack_b(const float* __restrict__ W, unsigned int* __restrict__ dst,
                         int N, int K) {
    int KC = K >> 5;
    int tid = blockIdx.x * blockDim.x + threadIdx.x;
    int total = (N * K) >> 1;
    if (tid >= total) return;
    int d    = tid & 7;
    int lane = (tid >> 3) & 31;
    int jc   = tid >> 8;
    int c    = jc % KC;
    int j    = jc / KC;
    int n    = j * 16 + (lane & 15);
    int kb   = c * 32 + ((lane >> 4) << 3) + d * 2 + ((d >= 4) ? 8 : 0);
    unsigned short lo = __builtin_bit_cast(unsigned short, (__bf16)W[(size_t)n * K + kb]);
    unsigned short hi = __builtin_bit_cast(unsigned short, (__bf16)W[(size_t)n * K + kb + 1]);
    dst[tid] = (unsigned int)lo | ((unsigned int)hi << 16);
}

// ---------- xp GEMM: xp[t][b][h] = x[b,t,:] . Whx[h,:] + bias[h], bf16 out ----------
__global__ __launch_bounds__(256) void xp_gemm(const float* __restrict__ x,
                                               const unsigned int* __restrict__ WhxB,
                                               const float* __restrict__ bias,
                                               __bf16* __restrict__ xp) {
    const int w     = threadIdx.x >> 5;
    const int lane  = threadIdx.x & 31;
    const int mtile = blockIdx.x;                // BT/16 tiles
    const int mrow  = lane & 15;
    const int hf    = lane >> 4;
    const size_t r  = (size_t)mtile * 16 + mrow; // BT row for A loads
    f32x8 acc[4] = {};
    #pragma unroll
    for (int kc = 0; kc < I_ / 32; ++kc) {
        const float* ap = x + r * I_ + kc * 32 + hf * 8;
        float4 a0 = *(const float4*)ap;
        float4 a1 = *(const float4*)(ap + 4);
        float4 a2 = *(const float4*)(ap + 16);
        float4 a3 = *(const float4*)(ap + 20);
        float fv[16] = {a0.x,a0.y,a0.z,a0.w, a1.x,a1.y,a1.z,a1.w,
                        a2.x,a2.y,a2.z,a2.w, a3.x,a3.y,a3.z,a3.w};
        bf16x16 af;
        #pragma unroll
        for (int i = 0; i < 16; ++i) af[i] = (__bf16)fv[i];
        #pragma unroll
        for (int j4 = 0; j4 < 4; ++j4) {
            int jt = w * 4 + j4;
            const u32x8* bp = (const u32x8*)(WhxB + ((size_t)(jt * (I_/32) + kc) * 32 + lane) * 8);
            bf16x16 bfr = __builtin_bit_cast(bf16x16, *bp);
            acc[j4] = wmma_bf16(af, bfr, acc[j4]);
        }
    }
    const int n = lane & 15;
    const int mbase = hf * 8;
    #pragma unroll
    for (int j4 = 0; j4 < 4; ++j4) {
        int jt = w * 4 + j4;
        float bv = bias[jt * 16 + n];
        #pragma unroll
        for (int v = 0; v < 8; ++v) {
            size_t rr = (size_t)mtile * 16 + mbase + v;   // BT row = b*T + t
            size_t b  = rr >> 10;
            size_t t  = rr & 1023;
            xp[(t * B_ + b) * H_ + jt * 16 + n] = (__bf16)(acc[j4][v] + bv);
        }
    }
}

// ---------- sequential scan + output projection ----------
// One WG per 16 batch rows; h in double-buffered LDS; Whh kc<KRES pinned in
// VGPRs for the whole scan, kc>=KRES streamed from L2 each step.
__global__ __launch_bounds__(256) void rnn_scan(const __bf16* __restrict__ xp,
                                                const unsigned int* __restrict__ WhhB,
                                                const unsigned int* __restrict__ WphB,
                                                const float* __restrict__ pbias,
                                                float* __restrict__ out) {
    const int w     = threadIdx.x >> 5;
    const int lane  = threadIdx.x & 31;
    const int bbase = blockIdx.x * 16;
    const int mrow  = lane & 15;
    const int hf    = lane >> 4;
    const int n     = lane & 15;
    const int mbase = hf * 8;

    __shared__ __align__(16) __bf16 hbuf[2][16 * RS];
    for (int i = threadIdx.x; i < 16 * RS; i += 256) hbuf[0][i] = (__bf16)0.0f;

    // Pin Whh B-fragments for kc = 0..KRES-1 in registers for the whole scan:
    // after each barrier the kc<KRES WMMAs can issue with zero load latency
    // while the kc>=KRES streaming loads are still in flight.
    bf16x16 bres[KRES][4];
    #pragma unroll
    for (int kc = 0; kc < KRES; ++kc)
        #pragma unroll
        for (int j4 = 0; j4 < 4; ++j4) {
            int jt = w * 4 + j4;
            bres[kc][j4] = __builtin_bit_cast(bf16x16,
                *(const u32x8*)(WhhB + ((size_t)(jt * 16 + kc) * 32 + lane) * 8));
        }

    // xp pipeline: xv holds step t, loaded one iteration ahead
    float xv[4][8];
    #pragma unroll
    for (int j4 = 0; j4 < 4; ++j4) {
        const __bf16* xq = xp + ((size_t)(bbase + mbase)) * H_ + (w * 4 + j4) * 16 + n;
        #pragma unroll
        for (int v = 0; v < 8; ++v) xv[j4][v] = (float)xq[(size_t)v * H_];
    }
    __syncthreads();

    for (int t = 0; t < T_; ++t) {
        asm volatile("" ::: "memory");   // keep streamed Whh loads inside the loop
        const __bf16* hc = hbuf[t & 1];
        __bf16*       hn = hbuf[(t + 1) & 1];

        // prefetch xp tile for step t+1 (consumed at next iteration's tanh)
        const int tn = (t + 1 < T_) ? (t + 1) : t;
        float xnx[4][8];
        #pragma unroll
        for (int j4 = 0; j4 < 4; ++j4) {
            const __bf16* xq = xp + ((size_t)tn * B_ + bbase + mbase) * H_ + (w * 4 + j4) * 16 + n;
            #pragma unroll
            for (int v = 0; v < 8; ++v) xnx[j4][v] = (float)xq[(size_t)v * H_];
        }

        f32x8 acc[4] = {};
        #pragma unroll
        for (int kc = 0; kc < KRES; ++kc) {          // zero-latency start
            bf16x16 af = load_a_lds(hc + mrow * RS + kc * 32 + hf * 8);
            #pragma unroll
            for (int j4 = 0; j4 < 4; ++j4)
                acc[j4] = wmma_bf16(af, bres[kc][j4], acc[j4]);
        }
        #pragma unroll
        for (int kc = KRES; kc < H_ / 32; ++kc) {    // streamed from L2
            bf16x16 af = load_a_lds(hc + mrow * RS + kc * 32 + hf * 8);
            #pragma unroll
            for (int j4 = 0; j4 < 4; ++j4) {
                int jt = w * 4 + j4;
                bf16x16 bfr = __builtin_bit_cast(bf16x16,
                    *(const u32x8*)(WhhB + ((size_t)(jt * 16 + kc) * 32 + lane) * 8));
                acc[j4] = wmma_bf16(af, bfr, acc[j4]);
            }
        }
        // tanh + write h_{t+1}
        #pragma unroll
        for (int j4 = 0; j4 < 4; ++j4) {
            int jt = w * 4 + j4;
            #pragma unroll
            for (int v = 0; v < 8; ++v) {
                float hv = fast_tanh(acc[j4][v] + xv[j4][v]);
                hn[(mbase + v) * RS + jt * 16 + n] = (__bf16)hv;
            }
        }
        // rotate xp pipeline
        #pragma unroll
        for (int j4 = 0; j4 < 4; ++j4)
            #pragma unroll
            for (int v = 0; v < 8; ++v) xv[j4][v] = xnx[j4][v];
        __syncthreads();
    }

    // out = h_T @ Wph^T + b ; h_T is in hbuf[0] (T_ is even)
    const __bf16* hT = hbuf[0];
    #pragma unroll
    for (int j2 = 0; j2 < 2; ++j2) {
        int jt = w * 2 + j2;                 // 16 O-tiles over 8 waves
        f32x8 acc = {};
        #pragma unroll
        for (int kc = 0; kc < H_ / 32; ++kc) {
            bf16x16 af = load_a_lds(hT + mrow * RS + kc * 32 + hf * 8);
            const u32x8* bp = (const u32x8*)(WphB + ((size_t)(jt * 16 + kc) * 32 + lane) * 8);
            bf16x16 bfr = __builtin_bit_cast(bf16x16, *bp);
            acc = wmma_bf16(af, bfr, acc);
        }
        float bv = pbias[jt * 16 + n];
        #pragma unroll
        for (int v = 0; v < 8; ++v)
            out[(size_t)(bbase + mbase + v) * O_ + jt * 16 + n] = acc[v] + bv;
    }
}

extern "C" void kernel_launch(void* const* d_in, const int* in_sizes, int n_in,
                              void* d_out, int out_size, void* d_ws, size_t ws_size,
                              hipStream_t stream) {
    const float* x     = (const float*)d_in[0];
    const float* Whx_w = (const float*)d_in[1];
    const float* Whx_b = (const float*)d_in[2];
    const float* Whh_w = (const float*)d_in[3];
    const float* Wph_w = (const float*)d_in[4];
    const float* Wph_b = (const float*)d_in[5];
    float* out = (float*)d_out;

    char* ws = (char*)d_ws;
    const size_t XP_BYTES = (size_t)T_ * B_ * H_ * 2;      // 256 MiB bf16 xp
    __bf16*       xp   = (__bf16*)ws;
    unsigned int* WhhB = (unsigned int*)(ws + XP_BYTES);
    unsigned int* WhxB = (unsigned int*)(ws + XP_BYTES + (size_t)H_ * H_ * 2);
    unsigned int* WphB = (unsigned int*)(ws + XP_BYTES + (size_t)H_ * H_ * 2 + (size_t)H_ * I_ * 2);

    repack_b<<<(H_ * H_ / 2) / 256, 256, 0, stream>>>(Whh_w, WhhB, H_, H_);
    repack_b<<<(H_ * I_ / 2) / 256, 256, 0, stream>>>(Whx_w, WhxB, H_, I_);
    repack_b<<<(O_ * H_ / 2) / 256, 256, 0, stream>>>(Wph_w, WphB, O_, H_);
    xp_gemm<<<(B_ * T_) / 16, 256, 0, stream>>>(x, WhxB, Whx_b, xp);
    rnn_scan<<<B_ / 16, 256, 0, stream>>>(xp, WhhB, WphB, Wph_b, out);
}
